// GraphAttentionLayer_6708738916707
// MI455X (gfx1250) — matrix operand order repaired and checked by reading, not compiled, not run on hardware
//
#include <hip/hip_runtime.h>
#include <hip/hip_bf16.h>

#define N_NODES 8192
#define FIN 512
#define FOUT 256
#define GAT_ALPHA 0.2f

typedef __attribute__((ext_vector_type(16))) __bf16 v16bf;
typedef __attribute__((ext_vector_type(8)))  float  v8f;
typedef __attribute__((ext_vector_type(8)))  __bf16 bf16x8;
typedef __attribute__((ext_vector_type(4)))  __bf16 bf16x4;
typedef __attribute__((ext_vector_type(4)))  float  f32x4;
typedef __attribute__((ext_vector_type(4)))  int    i32x4;

// ---------------------------------------------------------------- K1: h -> bf16
__global__ void cvt_h_kernel(const float* __restrict__ h, __bf16* __restrict__ hb) {
    int idx = (blockIdx.x * blockDim.x + threadIdx.x) * 4;
    f32x4 v = *(const f32x4*)(h + idx);
    bf16x4 o;
    o[0] = (__bf16)v[0]; o[1] = (__bf16)v[1]; o[2] = (__bf16)v[2]; o[3] = (__bf16)v[3];
    *(bf16x4*)(hb + idx) = o;
}

// ---------------------------------------------------------------- K2: W [512,256] f32 -> Wt [256,512] bf16
__global__ void transpose_W_kernel(const float* __restrict__ W, __bf16* __restrict__ Wt) {
    __shared__ float t[32][33];
    int bx = blockIdx.x * 32;   // column of W  (row of Wt)
    int by = blockIdx.y * 32;   // row of W
    int x = threadIdx.x, y0 = threadIdx.y;
    #pragma unroll
    for (int yy = y0; yy < 32; yy += 8)
        t[yy][x] = W[(size_t)(by + yy) * FOUT + bx + x];
    __syncthreads();
    #pragma unroll
    for (int yy = y0; yy < 32; yy += 8)
        Wt[(size_t)(bx + yy) * FIN + by + x] = (__bf16)t[x][yy];
}

// ---------------------------------------------------------------- K3: Wh = h @ W  (bf16 WMMA, f32 accum)
// block = 8 waves = one 16-row tile; wave w owns output col tiles 2w, 2w+1
__global__ void __launch_bounds__(256) gemm1_kernel(const __bf16* __restrict__ hb,
                                                    const __bf16* __restrict__ Wt,
                                                    float* __restrict__ Wh) {
    const int lane = threadIdx.x & 31;
    const int wave = threadIdx.x >> 5;
    const int rowbase = blockIdx.x * 16;
    const int m  = lane & 15;
    const int ks = (lane < 16) ? 0 : 16;   // A-layout K sub-offset
    const int kb = (lane < 16) ? 0 : 16;   // B-layout K sub-offset
    const int t0 = wave * 2;

    const __bf16* hrow  = hb + (size_t)(rowbase + m) * FIN;
    const __bf16* brow0 = Wt + (size_t)((t0 + 0) * 16 + m) * FIN + kb;
    const __bf16* brow1 = Wt + (size_t)((t0 + 1) * 16 + m) * FIN + kb;

    v8f acc0 = {}, acc1 = {};

    #pragma unroll 4
    for (int k0 = 0; k0 < FIN; k0 += 32) {
        bf16x8 lo = *(const bf16x8*)(hrow + k0 + ks);
        bf16x8 hi = *(const bf16x8*)(hrow + k0 + 8 + ks);
        v16bf A;
        #pragma unroll
        for (int e = 0; e < 8; ++e) { A[e] = lo[e]; A[8 + e] = hi[e]; }

        v16bf B0 = *(const v16bf*)(brow0 + k0);
        v16bf B1 = *(const v16bf*)(brow1 + k0);
        acc0 = __builtin_amdgcn_wmma_f32_16x16x32_bf16(false, A, false, B0, (short)0, acc0, false, false);
        acc1 = __builtin_amdgcn_wmma_f32_16x16x32_bf16(false, A, false, B1, (short)0, acc1, false, false);
    }

    #pragma unroll
    for (int r = 0; r < 8; ++r) {
        int row = rowbase + r + ((lane < 16) ? 0 : 8);
        Wh[(size_t)row * FOUT + (t0 + 0) * 16 + m] = acc0[r];
        Wh[(size_t)row * FOUT + (t0 + 1) * 16 + m] = acc1[r];
    }
}

// ---------------------------------------------------------------- K4: Wh [8192,256] f32 -> WhT [256,8192] bf16
__global__ void transpose_Wh_kernel(const float* __restrict__ Wh, __bf16* __restrict__ WhT) {
    __shared__ float t[32][33];
    int bx = blockIdx.x * 32;   // column of Wh (row of WhT)
    int by = blockIdx.y * 32;   // row of Wh
    int x = threadIdx.x, y0 = threadIdx.y;
    #pragma unroll
    for (int yy = y0; yy < 32; yy += 8)
        t[yy][x] = Wh[(size_t)(by + yy) * FOUT + bx + x];
    __syncthreads();
    #pragma unroll
    for (int yy = y0; yy < 32; yy += 8)
        WhT[(size_t)(bx + yy) * N_NODES + by + x] = (__bf16)t[x][yy];
}

// ---------------------------------------------------------------- K5: f1 = Wh@a1, f2 = Wh@a2 (one wave per row)
__global__ void f1f2_kernel(const float* __restrict__ Wh, const float* __restrict__ a,
                            float* __restrict__ f1, float* __restrict__ f2) {
    const int lane = threadIdx.x & 31;
    const int wave = threadIdx.x >> 5;
    const int row  = blockIdx.x * 8 + wave;
    const float* wr = Wh + (size_t)row * FOUT + lane * 8;
    const float* a1 = a + lane * 8;
    const float* a2 = a + FOUT + lane * 8;
    float s1 = 0.f, s2 = 0.f;
    #pragma unroll
    for (int e = 0; e < 8; ++e) {
        float v = wr[e];
        s1 = fmaf(v, a1[e], s1);
        s2 = fmaf(v, a2[e], s2);
    }
    #pragma unroll
    for (int off = 16; off > 0; off >>= 1) {
        s1 += __shfl_xor(s1, off, 32);
        s2 += __shfl_xor(s2, off, 32);
    }
    if (lane == 0) { f1[row] = s1; f2[row] = s2; }
}

// ---------------------------------------------------------------- K6: masked row softmax stats (max, sum-exp)
// adj is a 256MB single-use stream: non-temporal loads keep it from thrashing L2.
__global__ void row_stats_kernel(const int* __restrict__ adj, const float* __restrict__ f1,
                                 const float* __restrict__ f2,
                                 float* __restrict__ rowmax, float* __restrict__ rowsum) {
    __shared__ float sm[256];
    __shared__ float sl[256];
    const int row = blockIdx.x;
    const int tid = threadIdx.x;
    const int* ar = adj + (size_t)row * N_NODES;
    const float f1v = f1[row];

    float m = -__builtin_inff();
    float l = 0.f;
    for (int c = tid; c < N_NODES; c += 256) {
        int av = __builtin_nontemporal_load(ar + c);
        if (av > 0) {
            float e = f1v + f2[c];
            e = (e > 0.f) ? e : GAT_ALPHA * e;
            if (e > m) { l = l * __expf(m - e) + 1.0f; m = e; }
            else       { l += __expf(e - m); }
        }
    }
    sm[tid] = m; sl[tid] = l;
    for (int s = 128; s > 0; s >>= 1) {
        __syncthreads();
        if (tid < s) {
            float m1 = sm[tid], l1 = sl[tid];
            float m2 = sm[tid + s], l2 = sl[tid + s];
            float mm = fmaxf(m1, m2);
            float ll = 0.f;
            if (l1 > 0.f) ll += l1 * __expf(m1 - mm);
            if (l2 > 0.f) ll += l2 * __expf(m2 - mm);
            sm[tid] = mm; sl[tid] = ll;
        }
    }
    if (tid == 0) { rowmax[row] = sm[0]; rowsum[row] = sl[0]; }
}

// ---------------------------------------------------------------- K7: h_prime = softmax(mask(e)) @ Wh, ELU epilogue
// block = 8 waves = one 16-row tile x all 256 output cols; 64-wide K steps.
// Producer phase: 256 threads compute the 16x64 probability tile (4 entries each from an
// int4 adj load, non-temporal), staged as bf16 in LDS (double-buffered, row stride 72
// -> 16B-aligned chunks, conflict-free ds_load_b128).
// Consumer phase: wave w owns output col tiles 2w, 2w+1 (16 acc VGPRs); per barrier:
// 4 A-fragment ds_load_b128, 8 B-tile b128 loads (WhT, L2-resident), 4 WMMAs.
__global__ void __launch_bounds__(256) attn_gemm_kernel(const int* __restrict__ adj,
                                                        const __bf16* __restrict__ WhT,
                                                        const float* __restrict__ f1,
                                                        const float* __restrict__ f2,
                                                        const float* __restrict__ rowmax,
                                                        const float* __restrict__ rowsum,
                                                        float* __restrict__ out) {
    __shared__ __bf16 P[2][16][72];   // row stride 144B = 36 dwords: conflict-free b128 reads

    const int tid  = threadIdx.x;
    const int lane = tid & 31;
    const int wave = tid >> 5;
    const int rowbase = blockIdx.x * 16;

    // --- producer coordinates: thread handles P[sm][sk .. sk+3]
    const int sm_ = tid >> 4;          // 0..15
    const int sk  = (tid & 15) * 4;    // 0,4,..,60
    const int srow = rowbase + sm_;
    const float f1v  = f1[srow];
    const float mrow = rowmax[srow];
    const float lr   = rowsum[srow];
    const float linv = (lr > 0.f) ? (1.0f / lr) : 0.0f;
    const int*   adjp = adj + (size_t)srow * N_NODES + sk;
    const float* f2p  = f2 + sk;

    // --- consumer coordinates
    const int cm = lane & 15;
    const int ks = (lane < 16) ? 0 : 16;   // A-layout K sub-offset
    const int kb = (lane < 16) ? 0 : 16;   // B-layout K sub-offset
    const int t0 = wave * 2;
    const __bf16* brow0 = WhT + (size_t)((t0 + 0) * 16 + cm) * N_NODES + kb;
    const __bf16* brow1 = WhT + (size_t)((t0 + 1) * 16 + cm) * N_NODES + kb;

    v8f acc0 = {}, acc1 = {};

    for (int j = 0; j < N_NODES; j += 64) {
        const int buf = (j >> 6) & 1;

        // producer: masked softmax probabilities for this 16x64 tile (adj non-temporal)
        i32x4 av = __builtin_nontemporal_load((const i32x4*)(adjp + j));
        f32x4 fv = *(const f32x4*)(f2p + j);
        bf16x4 pp;
        #pragma unroll
        for (int q = 0; q < 4; ++q) {
            float e = f1v + fv[q];
            e = (e > 0.f) ? e : GAT_ALPHA * e;
            float p = (av[q] > 0) ? (__expf(e - mrow) * linv) : 0.0f;
            pp[q] = (__bf16)p;
        }
        *(bf16x4*)&P[buf][sm_][sk] = pp;   // single ds_store_b64

        __syncthreads();

        // consumer: two 32-wide WMMA K-substeps per barrier
        #pragma unroll
        for (int k0 = 0; k0 < 64; k0 += 32) {
            bf16x8 alo = *(const bf16x8*)&P[buf][cm][k0 + ks];
            bf16x8 ahi = *(const bf16x8*)&P[buf][cm][k0 + 8 + ks];
            v16bf A;
            #pragma unroll
            for (int e = 0; e < 8; ++e) { A[e] = alo[e]; A[8 + e] = ahi[e]; }

            v16bf B0 = *(const v16bf*)(brow0 + j + k0);
            v16bf B1 = *(const v16bf*)(brow1 + j + k0);
            acc0 = __builtin_amdgcn_wmma_f32_16x16x32_bf16(false, A, false, B0, (short)0, acc0, false, false);
            acc1 = __builtin_amdgcn_wmma_f32_16x16x32_bf16(false, A, false, B1, (short)0, acc1, false, false);
        }
    }

    // epilogue: ELU + non-temporal store (write-once output)
    #pragma unroll
    for (int r = 0; r < 8; ++r) {
        int orow = rowbase + r + ((lane < 16) ? 0 : 8);
        float x0 = acc0[r];
        float x1 = acc1[r];
        x0 = (x0 > 0.f) ? x0 : (__expf(x0) - 1.0f);
        x1 = (x1 > 0.f) ? x1 : (__expf(x1) - 1.0f);
        __builtin_nontemporal_store(x0, out + (size_t)orow * FOUT + (t0 + 0) * 16 + cm);
        __builtin_nontemporal_store(x1, out + (size_t)orow * FOUT + (t0 + 1) * 16 + cm);
    }
}

// ---------------------------------------------------------------- host launcher
extern "C" void kernel_launch(void* const* d_in, const int* in_sizes, int n_in,
                              void* d_out, int out_size, void* d_ws, size_t ws_size,
                              hipStream_t stream) {
    const float* h   = (const float*)d_in[0];   // [8192, 512]
    const int*   adj = (const int*)d_in[1];     // [8192, 8192]
    const float* W   = (const float*)d_in[2];   // [512, 256]
    const float* a   = (const float*)d_in[3];   // [512, 1]
    float* out = (float*)d_out;                 // [8192, 256]

    char* ws = (char*)d_ws;
    size_t off = 0;
    __bf16* hb   = (__bf16*)(ws + off); off += (size_t)N_NODES * FIN * 2;      // 8 MB
    __bf16* Wt   = (__bf16*)(ws + off); off += (size_t)FOUT * FIN * 2;         // 256 KB
    float*  Wh   = (float*)(ws + off);  off += (size_t)N_NODES * FOUT * 4;     // 8 MB
    __bf16* WhT  = (__bf16*)(ws + off); off += (size_t)FOUT * N_NODES * 2;     // 4 MB
    float*  f1   = (float*)(ws + off);  off += (size_t)N_NODES * 4;
    float*  f2   = (float*)(ws + off);  off += (size_t)N_NODES * 4;
    float*  rmax = (float*)(ws + off);  off += (size_t)N_NODES * 4;
    float*  rsum = (float*)(ws + off);  off += (size_t)N_NODES * 4;

    // K1: h -> bf16 (4M elements, 4 per thread)
    cvt_h_kernel<<<(N_NODES * FIN) / (256 * 4), 256, 0, stream>>>(h, hb);
    // K2: W -> Wt bf16 (transpose)
    transpose_W_kernel<<<dim3(FOUT / 32, FIN / 32), dim3(32, 8), 0, stream>>>(W, Wt);
    // K3: Wh = h @ W via bf16 WMMA (one block per 16-row tile)
    gemm1_kernel<<<N_NODES / 16, 256, 0, stream>>>(hb, Wt, Wh);
    // K4: Wh -> WhT bf16 (transpose)
    transpose_Wh_kernel<<<dim3(FOUT / 32, N_NODES / 32), dim3(32, 8), 0, stream>>>(Wh, WhT);
    // K5: f1/f2 per-row dot products (one wave per row)
    f1f2_kernel<<<N_NODES / 8, 256, 0, stream>>>(Wh, a, f1, f2);
    // K6: masked softmax row stats (one block per row)
    row_stats_kernel<<<N_NODES, 256, 0, stream>>>(adj, f1, f2, rmax, rsum);
    // K7: fused masked softmax x Wh via bf16 WMMA + ELU (one block per 16-row tile)
    attn_gemm_kernel<<<N_NODES / 16, 256, 0, stream>>>(adj, WhT, f1, f2, rmax, rsum, out);
}
